// GNNFiLMIK_53274774339888
// MI455X (gfx1250) — matrix operand order repaired and checked by reading, not compiled, not run on hardware
//
#include <hip/hip_runtime.h>
#include <math.h>

// ---------------------------------------------------------------------------
// MI455X (gfx1250) implementation. Compute-bound problem (~560 GFLOP vs a few
// hundred MB of traffic at 23.3 TB/s) -> all large GEMMs routed through
// v_wmma_f32_16x16x32_bf16 (bf16 operands, f32 accumulate).
// Conv2+mean uses t-major tiling: each 16-row WMMA tile = 16 samples at one
// timestep, so the time-mean is a register-level reduction (no LDS pass).
// K-loops are capped at unroll 2 and VGPRs at 256 (launch_bounds min-waves=4)
// so the compiler cannot spill or fall into the s_set_vgpr_msb high-VGPR path.
// ---------------------------------------------------------------------------

#define B_SZ 32768
#define H_SZ 256

typedef __bf16 bf_t;
typedef __attribute__((ext_vector_type(16))) __bf16 v16bf;
typedef __attribute__((ext_vector_type(8)))  float  v8f;

static __device__ __forceinline__ bf_t f2bf(float f) { return (bf_t)f; }

// Load a v16bf WMMA operand as two 16-byte chunks at element offsets 0 and e1.
static __device__ __forceinline__ v16bf ld_pair(const bf_t* p, int e1) {
  union { uint4 q[2]; v16bf v; } u;
  u.q[0] = *reinterpret_cast<const uint4*>(p);
  u.q[1] = *reinterpret_cast<const uint4*>(p + e1);
  return u.v;
}

static __device__ __forceinline__ v8f wmma_bf16(v16bf a, v16bf b, v8f c) {
  return __builtin_amdgcn_wmma_f32_16x16x32_bf16(false, a, false, b,
                                                 (short)0, c, false, false);
}

// ---------------------------------------------------------------------------
// Weight prep
// ---------------------------------------------------------------------------

// conv2 weight (256,256,3) f32 (o,i,k) -> bf16 (256, 768) with kk = k*256+i
__global__ __launch_bounds__(256) void prep_w2(const float* __restrict__ w2,
                                               bf_t* __restrict__ w2r) {
  int idx = blockIdx.x * 256 + threadIdx.x;      // 0..196607
  int o  = idx / 768;
  int kk = idx - o * 768;
  int k  = kk >> 8;
  int i  = kk & 255;
  w2r[idx] = f2bf(w2[(o * 256 + i) * 3 + k]);
}

__global__ __launch_bounds__(256) void cvt_bf(const float* __restrict__ s,
                                              bf_t* __restrict__ d, int n) {
  int i = blockIdx.x * 256 + threadIdx.x;
  if (i < n) d[i] = f2bf(s[i]);
}

// ---------------------------------------------------------------------------
// Temporal stage 1: conv1d (cin -> 256, k=3, pad=1) + relu, written as bf16
// h1p[b, tp, h], tp in [0,12), tp==0 / tp==11 are zero pad rows.
// ---------------------------------------------------------------------------
__global__ __launch_bounds__(256) void conv1_relu(
    const float* __restrict__ hf,    // (B,10,7)
    const float* __restrict__ w1,    // (256, cin, 3)
    const float* __restrict__ b1,    // (256)
    bf_t* __restrict__ h1p,          // (B,12,256)
    int c0, int cin) {
  int bx = blockIdx.x;               // b*12 + tp
  int b  = bx / 12;
  int tp = bx - b * 12;
  int h  = threadIdx.x;
  size_t oi = ((size_t)b * 12 + tp) * 256 + h;
  if (tp == 0 || tp == 11) { h1p[oi] = f2bf(0.0f); return; }
  int t = tp - 1;
  float acc = b1[h];
  const float* wr = w1 + h * cin * 3;
  for (int k = 0; k < 3; ++k) {
    int tt = t + k - 1;
    if (tt < 0 || tt >= 10) continue;
    const float* xr = hf + ((size_t)b * 10 + tt) * 7 + c0;
    for (int i = 0; i < cin; ++i) acc += wr[i * 3 + k] * xr[i];
  }
  h1p[oi] = f2bf(acc > 0.0f ? acc : 0.0f);
}

// ---------------------------------------------------------------------------
// Temporal stage 2: WMMA GEMM, t-major tiling. Each wave owns 16 samples x
// 32 output channels; for each t-group of 5 timesteps it keeps 10 f32
// accumulator tiles, then folds bias+relu into a register-level running mean.
// Per K-step: 10 A loads + 4 B loads for 10 WMMAs (1.4 loads/WMMA).
// Grid: (B/128, 8); block = 8 waves (each wave -> distinct 16 samples).
// ---------------------------------------------------------------------------
__global__ __launch_bounds__(256, 4) void conv2_mean_wmma(
    const bf_t*  __restrict__ h1p,   // (B,12,256)
    const bf_t*  __restrict__ w2r,   // (256,768)
    const float* __restrict__ b2,    // (256)
    float* __restrict__ nodef,       // (B,256)
    bf_t*  __restrict__ nodebf) {    // (B,256)
  const int tid   = threadIdx.x;
  const int lane  = tid & 31;
  const int wv    = tid >> 5;          // 0..7
  const int laneM = lane & 15;
  const int hi    = lane >> 4;         // 0/1 lane half
  const int n0    = blockIdx.y * 32;   // 32 output channels per block
  const int b0    = blockIdx.x * 128 + wv * 16;  // 16 samples per wave

  const bf_t* abase = h1p + (size_t)(b0 + laneM) * 12 * 256;  // sample row base
  const bf_t* brow0 = w2r + (size_t)(n0 + laneM) * 768;
  const bf_t* brow1 = w2r + (size_t)(n0 + 16 + laneM) * 768;
  const float bv0 = b2[n0 + laneM];
  const float bv1 = b2[n0 + 16 + laneM];

  v8f msum0 = {}, msum1 = {};
#pragma unroll 1
  for (int tg = 0; tg < 2; ++tg) {
    v8f acc[5][2] = {};
#pragma unroll 2
    for (int kk = 0; kk < 768; kk += 32) {
      v16bf bm0 = ld_pair(brow0 + kk + hi * 8, 16);
      v16bf bm1 = ld_pair(brow1 + kk + hi * 8, 16);
#pragma unroll
      for (int tt = 0; tt < 5; ++tt) {
        const bf_t* ar = abase + (size_t)(tg * 5 + tt) * 256 + kk + hi * 8;
        v16bf am = ld_pair(ar, 16);
        acc[tt][0] = wmma_bf16(am, bm0, acc[tt][0]);
        acc[tt][1] = wmma_bf16(am, bm1, acc[tt][1]);
      }
    }
#pragma unroll
    for (int tt = 0; tt < 5; ++tt) {
#pragma unroll
      for (int r = 0; r < 8; ++r) {
        float y0 = acc[tt][0][r] + bv0;
        float y1 = acc[tt][1][r] + bv1;
        msum0[r] += y0 > 0.0f ? y0 : 0.0f;
        msum1[r] += y1 > 0.0f ? y1 : 0.0f;
      }
    }
  }
#pragma unroll
  for (int r = 0; r < 8; ++r) {
    size_t gi0 = (size_t)(b0 + r + hi * 8) * 256 + n0 + laneM;
    float v0 = msum0[r] * 0.1f;
    float v1 = msum1[r] * 0.1f;
    nodef[gi0]       = v0;
    nodebf[gi0]      = f2bf(v0);
    nodef[gi0 + 16]  = v1;
    nodebf[gi0 + 16] = f2bf(v1);
  }
}

// ---------------------------------------------------------------------------
// Msg-passing linear: Y = X @ W^T + b, X = [xa | xb] (K = 256 or 512), N=256.
// MODE 0: out = old + Y (s2e).  MODE 1: g=sigmoid(Y); out = old + g*(Y-old).
// Wave = 4 M-tiles x 2 N-tiles (12 loads / 8 WMMAs). Block covers M=64, N=256.
// ---------------------------------------------------------------------------
template <int MODE>
__global__ __launch_bounds__(256, 4) void linear_wmma(
    const bf_t*  __restrict__ xa,    // (B,256)
    const bf_t*  __restrict__ xb,    // (B,256) or null
    const bf_t*  __restrict__ w,     // (256, K)
    const float* __restrict__ bias,  // (256)
    const float* __restrict__ oldf,  // (B,256)
    float* __restrict__ outf,        // (B,256)
    bf_t*  __restrict__ outbf,       // (B,256)
    int K) {
  const int tid   = threadIdx.x;
  const int lane  = tid & 31;
  const int wv    = tid >> 5;
  const int laneM = lane & 15;
  const int hi    = lane >> 4;
  const int n0    = wv * 32;                 // 8 waves x 32 cols = 256
  const int m0    = blockIdx.x * 64;

  const bf_t* brow0 = w + (size_t)(n0 + laneM) * K;
  const bf_t* brow1 = w + (size_t)(n0 + 16 + laneM) * K;
  v8f acc[4][2] = {};
#pragma unroll 2
  for (int kk = 0; kk < K; kk += 32) {
    v16bf bm0 = ld_pair(brow0 + kk + hi * 8, 16);
    v16bf bm1 = ld_pair(brow1 + kk + hi * 8, 16);
    const bf_t* src  = (kk < 256) ? xa : xb;
    int         kloc = (kk < 256) ? kk : (kk - 256);
#pragma unroll
    for (int mt = 0; mt < 4; ++mt) {
      const bf_t* ar = src + (size_t)(m0 + mt * 16 + laneM) * 256 + kloc + hi * 8;
      v16bf am = ld_pair(ar, 16);
      acc[mt][0] = wmma_bf16(am, bm0, acc[mt][0]);
      acc[mt][1] = wmma_bf16(am, bm1, acc[mt][1]);
    }
  }

  const float bv0 = bias[n0 + laneM];
  const float bv1 = bias[n0 + 16 + laneM];
#pragma unroll
  for (int mt = 0; mt < 4; ++mt) {
#pragma unroll
    for (int r = 0; r < 8; ++r) {
      int m = m0 + mt * 16 + r + hi * 8;
      size_t gi0 = (size_t)m * 256 + n0 + laneM;
#pragma unroll
      for (int cg = 0; cg < 2; ++cg) {
        size_t gi = gi0 + cg * 16;
        float y = acc[mt][cg][r] + (cg ? bv1 : bv0);
        float v;
        if (MODE == 0) {
          v = oldf[gi] + y;
        } else {
          float g  = 1.0f / (1.0f + expf(-y));
          float ov = oldf[gi];
          v = ov + g * (y - ov);
        }
        outf[gi]  = v;
        outbf[gi] = f2bf(v);
      }
    }
  }
}

// ---------------------------------------------------------------------------
// FiLM + endpose: one wave per sample. Wave32 shuffle reductions for LN,
// exact-erf GELU, then endpose MLPs; adds endeff into wrist node.
// Branch order used here: 0=pos_gamma, 1=pos_beta, 2=ori_gamma, 3=ori_beta.
// ---------------------------------------------------------------------------
struct FilmParams {
  const float* hf;
  const float* ep;
  const float* eo;
  const float* l1w[4]; const float* l1b[4];
  const float* lng[4]; const float* lnb[4];
  const float* l2w[4]; const float* l2b[4];
  const float *pw1, *pb1, *pw2, *pb2;   // endpose pos: lin(128,3), lin(256,128)
  const float *ow1, *ob1, *ow2, *ob2;   // endpose ori: lin(128,4), lin(256,128)
  float* wristf;
  bf_t*  wristbf;
};

__global__ __launch_bounds__(256) void film_endpose(FilmParams P) {
  __shared__ float h128s[8][2][128];
  const int tid  = threadIdx.x;
  const int lane = tid & 31;
  const int wv   = tid >> 5;
  const int b    = blockIdx.x * 8 + wv;

  float cur[7];
#pragma unroll
  for (int j = 0; j < 7; ++j) cur[j] = P.hf[((size_t)b * 10 + 9) * 7 + j];
  float epv[3], eov[4];
#pragma unroll
  for (int j = 0; j < 3; ++j) epv[j] = P.ep[b * 3 + j];
#pragma unroll
  for (int j = 0; j < 4; ++j) eov[j] = P.eo[b * 4 + j];

  const int douts[4] = {3, 3, 4, 4};
  float bro[4][4];
  for (int br = 0; br < 4; ++br) {
    float hreg[8];
    float s1 = 0.0f, s2 = 0.0f;
#pragma unroll
    for (int q = 0; q < 8; ++q) {
      int idx = lane + q * 32;
      const float* wr = P.l1w[br] + idx * 7;
      float a = P.l1b[br][idx];
#pragma unroll
      for (int j = 0; j < 7; ++j) a += wr[j] * cur[j];
      hreg[q] = a; s1 += a; s2 += a * a;
    }
    for (int off = 16; off; off >>= 1) {
      s1 += __shfl_xor(s1, off, 32);
      s2 += __shfl_xor(s2, off, 32);
    }
    float mu  = s1 * (1.0f / 256.0f);
    float var = s2 * (1.0f / 256.0f) - mu * mu;
    float inv = rsqrtf(var + 1e-5f);
    int d = douts[br];
    float part[4] = {0.0f, 0.0f, 0.0f, 0.0f};
#pragma unroll
    for (int q = 0; q < 8; ++q) {
      int idx = lane + q * 32;
      float xn = (hreg[q] - mu) * inv * P.lng[br][idx] + P.lnb[br][idx];
      float ge = 0.5f * xn * (1.0f + erff(xn * 0.70710678118654752f));
      for (int dd = 0; dd < d; ++dd) part[dd] += ge * P.l2w[br][dd * 256 + idx];
    }
    for (int off = 16; off; off >>= 1)
      for (int dd = 0; dd < 4; ++dd) part[dd] += __shfl_xor(part[dd], off, 32);
    for (int dd = 0; dd < d; ++dd) bro[br][dd] = part[dd] + P.l2b[br][dd];
  }

  float mp[3], mo[4];
#pragma unroll
  for (int j = 0; j < 3; ++j) mp[j] = bro[0][j] * epv[j] + bro[1][j];
#pragma unroll
  for (int j = 0; j < 4; ++j) mo[j] = bro[2][j] * eov[j] + bro[3][j];

#pragma unroll
  for (int q = 0; q < 4; ++q) {
    int idx = lane + q * 32;
    float a = P.pb1[idx];
#pragma unroll
    for (int j = 0; j < 3; ++j) a += P.pw1[idx * 3 + j] * mp[j];
    h128s[wv][0][idx] = a > 0.0f ? a : 0.0f;
    float a2 = P.ob1[idx];
#pragma unroll
    for (int j = 0; j < 4; ++j) a2 += P.ow1[idx * 4 + j] * mo[j];
    h128s[wv][1][idx] = a2 > 0.0f ? a2 : 0.0f;
  }
  __syncthreads();

#pragma unroll
  for (int q = 0; q < 8; ++q) {
    int o = lane + q * 32;
    float a = P.pb2[o] + P.ob2[o];
    const float* wp = P.pw2 + o * 128;
    const float* wo = P.ow2 + o * 128;
    for (int k = 0; k < 128; ++k)
      a += wp[k] * h128s[wv][0][k] + wo[k] * h128s[wv][1][k];
    size_t gi = (size_t)b * 256 + o;
    float v = P.wristf[gi] + a;
    P.wristf[gi]  = v;
    P.wristbf[gi] = f2bf(v);
  }
}

// ---------------------------------------------------------------------------
// Output heads: pred[b, 0:3]=shoulder, [3]=elbow, [4]=forearm, [5:7]=wrist.
// ---------------------------------------------------------------------------
__global__ __launch_bounds__(256) void heads_kernel(
    const float* __restrict__ sh, const float* __restrict__ el,
    const float* __restrict__ fo, const float* __restrict__ wr,
    const float* __restrict__ wsho, const float* __restrict__ bsho,
    const float* __restrict__ welb, const float* __restrict__ belb,
    const float* __restrict__ wfor, const float* __restrict__ bfor,
    const float* __restrict__ wwri, const float* __restrict__ bwri,
    float* __restrict__ out) {
  int g = blockIdx.x * 256 + threadIdx.x;  // b*8 + j
  int b = g >> 3, j = g & 7;
  if (j >= 7) return;
  const float* x; const float* w; float bias;
  if (j < 3)      { x = sh; w = wsho + j * 256;       bias = bsho[j];     }
  else if (j == 3){ x = el; w = welb;                 bias = belb[0];     }
  else if (j == 4){ x = fo; w = wfor;                 bias = bfor[0];     }
  else            { x = wr; w = wwri + (j - 5) * 256; bias = bwri[j - 5]; }
  const float4* xv = reinterpret_cast<const float4*>(x + (size_t)b * 256);
  const float4* wv = reinterpret_cast<const float4*>(w);
  float acc = bias;
  for (int k = 0; k < 64; ++k) {
    float4 a = xv[k], c = wv[k];
    acc += a.x * c.x + a.y * c.y + a.z * c.z + a.w * c.w;
  }
  out[b * 7 + j] = acc;
}

// ---------------------------------------------------------------------------
// Host launcher. Input indices assume jax-style pytree flattening:
// [history_frames, end_position, end_orientation] then params with dict keys
// sorted at every level and lists in order (each lin -> b, w; ln -> b, g).
// ---------------------------------------------------------------------------
extern "C" void kernel_launch(void* const* d_in, const int* in_sizes, int n_in,
                              void* d_out, int out_size, void* d_ws, size_t ws_size,
                              hipStream_t stream) {
  (void)in_sizes; (void)n_in; (void)out_size; (void)ws_size;
  const float* hf = (const float*)d_in[0];
  const float* ep = (const float*)d_in[1];
  const float* eo = (const float*)d_in[2];
  #define PF(i) ((const float*)d_in[(i)])

  // --- workspace carve-up (~387 MB total) ---
  char* ws = (char*)d_ws;
  size_t off = 0;
  bf_t* h1p = (bf_t*)(ws + off); off += (size_t)B_SZ * 12 * 256 * 2;     // 201.3 MB
  bf_t* w2r[4];
  for (int g = 0; g < 4; ++g) { w2r[g] = (bf_t*)(ws + off); off += 196608 * 2; }
  bf_t *s2e_bf[2], *tof_bf[2], *f2w_bf[2];
  for (int l = 0; l < 2; ++l) {
    s2e_bf[l] = (bf_t*)(ws + off); off += 65536 * 2;
    tof_bf[l] = (bf_t*)(ws + off); off += 131072 * 2;
    f2w_bf[l] = (bf_t*)(ws + off); off += 131072 * 2;
  }
  float* nodef[4];
  for (int g = 0; g < 4; ++g) { nodef[g] = (float*)(ws + off); off += (size_t)B_SZ * 256 * 4; }
  bf_t* nodebf[4];
  for (int g = 0; g < 4; ++g) { nodebf[g] = (bf_t*)(ws + off); off += (size_t)B_SZ * 256 * 2; }

  // --- param indices (see flattening note above) ---
  const int FILM_ORI_BETA = 11, FILM_ORI_GAMMA = 17, FILM_POS_BETA = 23, FILM_POS_GAMMA = 29;
  const int MSG_BASE[2] = {43, 49};                 // f2w.b,f2w.w,s2e.b,s2e.w,to_f.b,to_f.w
  // temporal: node order {shoulder,elbow,forearm,wrist} -> param bases
  const int T_BASE[4] = {63, 55, 59, 67};           // c1.b,c1.w,c2.b,c2.w
  const int C0[4]  = {0, 3, 4, 5};
  const int CIN[4] = {3, 1, 1, 2};

  // --- 1. weight prep ---
  for (int g = 0; g < 4; ++g)
    prep_w2<<<768, 256, 0, stream>>>(PF(T_BASE[g] + 3), w2r[g]);
  for (int l = 0; l < 2; ++l) {
    cvt_bf<<<(131072 + 255) / 256, 256, 0, stream>>>(PF(MSG_BASE[l] + 1), f2w_bf[l], 131072);
    cvt_bf<<<(65536  + 255) / 256, 256, 0, stream>>>(PF(MSG_BASE[l] + 3), s2e_bf[l], 65536);
    cvt_bf<<<(131072 + 255) / 256, 256, 0, stream>>>(PF(MSG_BASE[l] + 5), tof_bf[l], 131072);
  }

  // --- 2. temporal branches (heavy WMMA GEMMs) ---
  for (int g = 0; g < 4; ++g) {
    conv1_relu<<<B_SZ * 12, 256, 0, stream>>>(
        hf, PF(T_BASE[g] + 1), PF(T_BASE[g] + 0), h1p, C0[g], CIN[g]);
    conv2_mean_wmma<<<dim3(B_SZ / 128, 8), 256, 0, stream>>>(
        h1p, w2r[g], PF(T_BASE[g] + 2), nodef[g], nodebf[g]);
  }

  // --- 3. FiLM + endpose -> wrist ---
  FilmParams FP;
  FP.hf = hf; FP.ep = ep; FP.eo = eo;
  const int fb[4] = {FILM_POS_GAMMA, FILM_POS_BETA, FILM_ORI_GAMMA, FILM_ORI_BETA};
  for (int br = 0; br < 4; ++br) {
    FP.l1b[br] = PF(fb[br] + 0); FP.l1w[br] = PF(fb[br] + 1);
    FP.l2b[br] = PF(fb[br] + 2); FP.l2w[br] = PF(fb[br] + 3);
    FP.lnb[br] = PF(fb[br] + 4); FP.lng[br] = PF(fb[br] + 5);
  }
  FP.ob1 = PF(3);  FP.ow1 = PF(4);  FP.ob2 = PF(5);  FP.ow2 = PF(6);
  FP.pb1 = PF(7);  FP.pw1 = PF(8);  FP.pb2 = PF(9);  FP.pw2 = PF(10);
  FP.wristf = nodef[3]; FP.wristbf = nodebf[3];
  film_endpose<<<B_SZ / 8, 256, 0, stream>>>(FP);

  // --- 4. gated message passing (WMMA linears with fused epilogues) ---
  dim3 lgrid(B_SZ / 64);
  for (int l = 0; l < 2; ++l) {
    // elbow += shoulder @ s2e
    linear_wmma<0><<<lgrid, 256, 0, stream>>>(
        nodebf[0], (const bf_t*)nullptr, s2e_bf[l], PF(MSG_BASE[l] + 2),
        nodef[1], nodef[1], nodebf[1], 256);
    // forearm gate with fmsg([shoulder|elbow] @ to_f)
    linear_wmma<1><<<lgrid, 256, 0, stream>>>(
        nodebf[0], nodebf[1], tof_bf[l], PF(MSG_BASE[l] + 4),
        nodef[2], nodef[2], nodebf[2], 512);
    // wrist gate with wmsg([elbow|forearm] @ f2w)
    linear_wmma<1><<<lgrid, 256, 0, stream>>>(
        nodebf[1], nodebf[2], f2w_bf[l], PF(MSG_BASE[l] + 0),
        nodef[3], nodef[3], nodebf[3], 512);
  }

  // --- 5. heads ---
  heads_kernel<<<(B_SZ * 8) / 256, 256, 0, stream>>>(
      nodef[0], nodef[1], nodef[2], nodef[3],
      PF(40), PF(39),   // shoulder w,b
      PF(36), PF(35),   // elbow w,b
      PF(38), PF(37),   // forearm w,b
      PF(42), PF(41),   // wrist w,b
      (float*)d_out);
  #undef PF
}
